// Network_4655744548946
// MI455X (gfx1250) — matrix operand order repaired and checked by reading, not compiled, not run on hardware
//
#include <hip/hip_runtime.h>
#include <hip/hip_bf16.h>

typedef _Float16 v16h __attribute__((ext_vector_type(16)));
typedef _Float16 h8   __attribute__((ext_vector_type(8)));
typedef float    v8f  __attribute__((ext_vector_type(8)));

typedef int v4i __attribute__((vector_size(16)));
typedef __attribute__((address_space(3))) v4i lv4i;   /* LDS int4 */

#define NPTS    32768
#define LVLS    16
#define TSZ     (1u<<19)
#define KNP     3968          /* 120*32 + 32 + 32*3 */
#define NETS    48
#define ENT_CAP 101376        /* 3*NPTS + 48*64 padding headroom, mult of 64 */
#define WS_COUNTS 0
#define WS_CURS   64
#define WS_OFFS   128         /* 49 ints: padded bin offsets + total */
#define WS_ENT    256
#define NETIN_BYTE_OFF (512u*1024u)
#define W1H_BYTE_OFF   (9u*1024u*1024u)   /* 48 x [32][128] f16 = 384 KB */

__constant__ float cRES[16] = {16.f,21.f,27.f,36.f,48.f,64.f,84.f,111.f,
                               147.f,194.f,256.f,337.f,445.f,588.f,776.f,1024.f};

__global__ void init_kernel(float* out, int* wsI){
  int i = blockIdx.x*blockDim.x + threadIdx.x;
  if (i < NPTS*3)  out[i] = 0.f;
  if (i < ENT_CAP) wsI[WS_ENT + i] = -1;
  if (i < NETS)    { wsI[WS_COUNTS + i] = 0; wsI[WS_CURS + i] = 0; }
}

/* one-time: all 48 W1 matrices -> transposed f16 [h][k] with k=120..127 zeroed */
__global__ void w1cvt_kernel(const float* __restrict__ kn, _Float16* __restrict__ w1h){
  int net = blockIdx.x;
  const float* W1 = kn + (size_t)net*KNP;
  _Float16* dst = w1h + (size_t)net*4096;
  for (int g = threadIdx.x; g < 3840; g += blockDim.x){
    int K = g >> 5, h = g & 31;
    dst[h*128 + K] = (_Float16)W1[g];
  }
  for (int z = threadIdx.x; z < 256; z += blockDim.x){
    int h = z >> 3, K = 120 + (z & 7);
    dst[h*128 + K] = (_Float16)0.f;
  }
}

__global__ void encode_kernel(const float* __restrict__ norm,
                              const float* __restrict__ view,
                              const float* __restrict__ tptr,
                              const float* __restrict__ txy,
                              const float* __restrict__ txz,
                              const float* __restrict__ tyz,
                              int* wsI, _Float16* netin){
  int p = blockIdx.x*blockDim.x + threadIdx.x;
  if (p >= NPTS) return;
  float x0 = norm[p*3+0], x1 = norm[p*3+1], x2 = norm[p*3+2];
  float tv = tptr[0];
  _Float16* dst = netin + (size_t)p*128;
  const float* tabs[3] = {txy, txz, tyz};
  float c0s[3] = {x0, x0, x1};
  float c1s[3] = {x1, x2, x2};
  for (int pl = 0; pl < 3; ++pl){
    const float* tbl = tabs[pl];
    float a = c0s[pl], b = c1s[pl];
    for (int l = 0; l < LVLS; ++l){
      float r  = cRES[l], rm = r - 1.f;
      float p0 = fminf(fmaxf(a*r, 0.f), rm);
      float p1 = fminf(fmaxf(b*r, 0.f), rm);
      float p2 = fminf(fmaxf(tv*r,0.f), rm);
      float f0 = floorf(p0), f1 = floorf(p1), f2 = floorf(p2);
      float r0 = p0-f0, r1 = p1-f1, r2 = p2-f2;
      unsigned u0 = (unsigned)f0, u1 = (unsigned)f1, u2 = (unsigned)f2;
      float acc0 = 0.f, acc1 = 0.f;
      const float* base = tbl + (size_t)l*TSZ*2u;
      #pragma unroll
      for (int c = 0; c < 8; ++c){
        unsigned b0 = (c>>2)&1u, b1c = (c>>1)&1u, b2c = c&1u;
        unsigned hsh = (u0+b0) ^ ((u1+b1c)*2654435761u) ^ ((u2+b2c)*805459861u);
        unsigned idx = hsh & (TSZ-1u);
        float w = (b0 ? r0 : 1.f-r0)*(b1c ? r1 : 1.f-r1)*(b2c ? r2 : 1.f-r2);
        const float* fp = base + (size_t)idx*2u;
        acc0 += w*fp[0]; acc1 += w*fp[1];
      }
      union { _Float16 h[2]; unsigned u; } pk;
      pk.h[0] = (_Float16)acc0; pk.h[1] = (_Float16)acc1;
      *(unsigned*)(dst + pl*32 + l*2) = pk.u;   /* 4B aligned packed store */
    }
  }
  for (int j = 0; j < 3; ++j){
    float vj = view[p*3+j];
    h8 hv;
    #pragma unroll
    for (int f = 0; f < 4; ++f){
      float ang = vj * (float)(1<<f);
      hv[f]     = (_Float16)__sinf(ang);
      hv[4 + f] = (_Float16)__cosf(ang);
    }
    *(h8*)(dst + 96 + j*8) = hv;
  }
  { h8 z = {}; *(h8*)(dst + 120) = z; }
  int i0 = min(max((int)floorf(x0*4.f),0),3);
  int i1 = min(max((int)floorf(x1*4.f),0),3);
  int i2 = min(max((int)floorf(x2*4.f),0),3);
  atomicAdd(&wsI[WS_COUNTS +  0 + i0*4 + i1], 1);
  atomicAdd(&wsI[WS_COUNTS + 16 + i0*4 + i2], 1);
  atomicAdd(&wsI[WS_COUNTS + 32 + i1*4 + i2], 1);
}

__global__ void prefix_kernel(int* wsI){
  if (blockIdx.x == 0 && threadIdx.x == 0){
    int off = 0;
    for (int i = 0; i < NETS; ++i){
      wsI[WS_OFFS + i] = off;
      int cnt = wsI[WS_COUNTS + i];
      off += (cnt + 63) & ~63;          /* pad each bin to 64: 4 tiles/block share net */
    }
    wsI[WS_OFFS + NETS] = off;          /* total padded entries (mult of 64) */
  }
}

__global__ void scatter_kernel(const float* __restrict__ norm, int* wsI){
  int p = blockIdx.x*blockDim.x + threadIdx.x;
  if (p >= NPTS) return;
  float x0 = norm[p*3], x1 = norm[p*3+1], x2 = norm[p*3+2];
  int i0 = min(max((int)floorf(x0*4.f),0),3);
  int i1 = min(max((int)floorf(x1*4.f),0),3);
  int i2 = min(max((int)floorf(x2*4.f),0),3);
  int nets[3] = {0 + i0*4 + i1, 16 + i0*4 + i2, 32 + i1*4 + i2};
  for (int pl = 0; pl < 3; ++pl){
    int net = nets[pl];
    int pos = wsI[WS_OFFS + net] + atomicAdd(&wsI[WS_CURS + net], 1);
    wsI[WS_ENT + pos] = p;
  }
}

__global__ __launch_bounds__(128)
void gemm_kernel(const float* __restrict__ kn,
                 const _Float16* __restrict__ netin,
                 const _Float16* __restrict__ w1h,
                 const int* __restrict__ wsI,
                 float* out){
  __shared__ _Float16 Wt[32][128];       /* staged via async-to-LDS */
  __shared__ _Float16 h1s[4][16][32];    /* per-wave h1 staging */
  __shared__ float b1s[32];
  __shared__ float w2s[96];
  __shared__ int   sEnt[64];
  __shared__ int   sNet;

  int tb    = blockIdx.x * 64;
  int total = wsI[WS_OFFS + NETS];
  if (tb >= total) return;               /* block-uniform: EXEC stays full */
  int tid = threadIdx.x;

  if (tid == 0){
    int n = 0;
    while (n < NETS-1 && wsI[WS_OFFS + n + 1] <= tb) ++n;
    sNet = n;
  }
  if (tid < 64) sEnt[tid] = wsI[WS_ENT + tb + tid];
  __syncthreads();
  int net = sNet;

  const float* b1 = kn + (size_t)net*KNP + 3840;
  const float* W2 = b1 + 32;
  if (tid < 32) b1s[tid] = b1[tid];
  else          w2s[tid - 32] = W2[tid - 32];   /* tids 32..127 -> 96 floats */

  /* copy W1(net) f16-transposed [32][128] (8 KB) global -> LDS */
  {
    const char* gsrc = (const char*)(w1h + (size_t)net*4096);
    char* lbase = (char*)&Wt[0][0];
    unsigned off = (unsigned)tid * 64u;
#if __has_builtin(__builtin_amdgcn_global_load_async_to_lds_b128)
    #pragma unroll
    for (int i = 0; i < 4; ++i){
      __builtin_amdgcn_global_load_async_to_lds_b128(
          (v4i*)(gsrc + off + i*16),
          (lv4i*)(lbase + off + i*16),
          0, 0);
    }
  #if __has_builtin(__builtin_amdgcn_s_wait_asynccnt)
    __builtin_amdgcn_s_wait_asynccnt(0);
  #else
    asm volatile("s_wait_asynccnt 0x0" ::: "memory");
  #endif
#else
    #pragma unroll
    for (int i = 0; i < 4; ++i){
      h8 v = *(const h8*)(gsrc + off + i*16);
      *(h8*)(lbase + off + i*16) = v;
    }
#endif
  }
  __syncthreads();

  int wv = tid >> 5, lane = tid & 31;
  int m = lane & 15, hf = lane >> 4;
  int e = sEnt[wv*16 + m];
  const _Float16* arow = netin + (size_t)(e < 0 ? 0 : e) * 128;
  v8f acc0 = {}; v8f acc1 = {};

  #pragma unroll
  for (int c = 0; c < 4; ++c){
    v16h av;
    if (e >= 0){
      h8 lo = *(const h8*)(arow + c*32 + hf*8);
      h8 hi = *(const h8*)(arow + c*32 + 16 + hf*8);
      av = __builtin_shufflevector(lo, hi, 0,1,2,3,4,5,6,7,8,9,10,11,12,13,14,15);
    } else {
      v16h zz = {}; av = zz;
    }
    {
      const _Float16* bp = &Wt[m][c*32 + hf*8];
      h8 blo = *(const h8*)bp;
      h8 bhi = *(const h8*)(bp + 16);
      v16h bv = __builtin_shufflevector(blo, bhi, 0,1,2,3,4,5,6,7,8,9,10,11,12,13,14,15);
      acc0 = __builtin_amdgcn_wmma_f32_16x16x32_f16(false, av, false, bv,
                                                    (short)0, acc0, false, false);
    }
    {
      const _Float16* bp = &Wt[16 + m][c*32 + hf*8];
      h8 blo = *(const h8*)bp;
      h8 bhi = *(const h8*)(bp + 16);
      v16h bv = __builtin_shufflevector(blo, bhi, 0,1,2,3,4,5,6,7,8,9,10,11,12,13,14,15);
      acc1 = __builtin_amdgcn_wmma_f32_16x16x32_f16(false, av, false, bv,
                                                    (short)0, acc1, false, false);
    }
  }

  float bb0 = b1s[m], bb1 = b1s[16 + m];
  #pragma unroll
  for (int r = 0; r < 8; ++r){
    int row = hf ? 8 + r : r;
    float v0 = acc0[r] + bb0; v0 = v0 > 0.f ? v0 : 0.f;
    float v1 = acc1[r] + bb1; v1 = v1 > 0.f ? v1 : 0.f;
    h1s[wv][row][m]      = (_Float16)v0;
    h1s[wv][row][16 + m] = (_Float16)v1;
  }
  __syncthreads();

  h8 alo = *(const h8*)&h1s[wv][m][hf*8];
  h8 ahi = *(const h8*)&h1s[wv][m][16 + hf*8];
  v16h a2 = __builtin_shufflevector(alo, ahi, 0,1,2,3,4,5,6,7,8,9,10,11,12,13,14,15);
  v16h b2;
  #pragma unroll
  for (int e2 = 0; e2 < 16; ++e2){
    int K = (e2 < 8) ? hf*8 + e2 : 16 + hf*8 + (e2 - 8);
    b2[e2] = (m < 3) ? (_Float16)w2s[K*3 + m] : (_Float16)0.f;
  }
  v8f acc2 = {};
  acc2 = __builtin_amdgcn_wmma_f32_16x16x32_f16(false, a2, false, b2,
                                                (short)0, acc2, false, false);
  if (m < 3){
    #pragma unroll
    for (int r = 0; r < 8; ++r){
      int row = hf ? 8 + r : r;
      int ee = sEnt[wv*16 + row];
      if (ee >= 0) atomicAdd(out + ee*3 + m, acc2[r] * (1.f/3.f));
    }
  }
}

extern "C" void kernel_launch(void* const* d_in, const int* in_sizes, int n_in,
                              void* d_out, int out_size, void* d_ws, size_t ws_size,
                              hipStream_t stream){
  const float* norm = (const float*)d_in[0];
  const float* view = (const float*)d_in[1];
  const float* tp   = (const float*)d_in[2];
  const float* txy  = (const float*)d_in[3];
  const float* txz  = (const float*)d_in[4];
  const float* tyz  = (const float*)d_in[5];
  const float* kn   = (const float*)d_in[6];
  float* out = (float*)d_out;
  int* wsI = (int*)d_ws;
  _Float16* netin = (_Float16*)((char*)d_ws + NETIN_BYTE_OFF);
  _Float16* w1h   = (_Float16*)((char*)d_ws + W1H_BYTE_OFF);

  init_kernel   <<<(ENT_CAP+255)/256, 256, 0, stream>>>(out, wsI);
  w1cvt_kernel  <<<NETS,             128, 0, stream>>>(kn, w1h);
  encode_kernel <<<NPTS/256,         256, 0, stream>>>(norm, view, tp, txy, txz, tyz, wsI, netin);
  prefix_kernel <<<1,                 32, 0, stream>>>(wsI);
  scatter_kernel<<<NPTS/256,         256, 0, stream>>>(norm, wsI);
  gemm_kernel   <<<ENT_CAP/64,       128, 0, stream>>>(kn, netin, w1h, wsI, out);
}
